// MPGNN_33019708572043
// MI455X (gfx1250) — matrix operand order repaired and checked by reading, not compiled, not run on hardware
//
#include <hip/hip_runtime.h>

typedef __bf16 bf16_t;
typedef float        v8f   __attribute__((ext_vector_type(8)));
typedef bf16_t       v16bf __attribute__((ext_vector_type(16)));
typedef bf16_t       v8bf  __attribute__((ext_vector_type(8)));
typedef unsigned int u32x4 __attribute__((ext_vector_type(4)));
typedef int          i32x8 __attribute__((ext_vector_type(8)));
typedef int          i32x4 __attribute__((ext_vector_type(4)));

#define WPB 8                 // waves per block
#define TPB (WPB * 32)

__device__ __forceinline__ void wave_sync() {
  __builtin_amdgcn_wave_barrier();
  asm volatile("" ::: "memory");
}

__device__ __forceinline__ v8f wmma_bf16(v16bf a, v16bf b, v8f c) {
  return __builtin_amdgcn_wmma_f32_16x16x32_bf16(false, a, false, b, (short)0, c,
                                                 false, false);
}

__device__ __forceinline__ void atomicAddF32(float* p, float v) {
  unsafeAtomicAdd(p, v);  // -> global_atomic_add_f32
}

// ---- fill an LDS region from global (pure byte copy of pre-swizzled weights) ----
// TDM path: 1D tensor, data_size=8B, tile_dim0 = nbytes/8 (<=65535).
__device__ __forceinline__ void fill_lds_from_global(bf16_t* ldst, const bf16_t* gsrc,
                                                     unsigned int nbytes) {
#if __has_builtin(__builtin_amdgcn_tensor_load_to_lds)
  if ((threadIdx.x >> 5) == 0) {  // wave 0 issues the TDM op (EXEC ignored by TDM)
    unsigned long long ga = (unsigned long long)(uintptr_t)gsrc;
    unsigned int ldsoff = (unsigned int)(uintptr_t)ldst;  // low 32 bits = LDS offset
    unsigned int elems = nbytes >> 3;                     // 8-byte elements
    u32x4 g0;
    g0[0] = 1u;                                   // count=1, no gather, user mode
    g0[1] = ldsoff;                               // lds_addr
    g0[2] = (unsigned int)ga;                     // global_addr[31:0]
    g0[3] = (unsigned int)((ga >> 32) & 0x01FFFFFFu) | (2u << 30);  // [56:32] | type=2
    i32x8 g1;
    g1[0] = (int)(3u << 16);                      // data_size=3 (8B), no multicast
    g1[1] = (int)((elems & 0xFFFFu) << 16);       // tensor_dim0[15:0]
    g1[2] = (int)(((elems >> 16) & 0xFFFFu) | (1u << 16));  // t_dim0[31:16], t_dim1=1
    g1[3] = (int)((elems & 0xFFFFu) << 16);       // tile_dim0 = elems
    g1[4] = 0;                                    // tile_dim1/2 unused
    g1[5] = (int)elems;                           // tensor_dim0_stride[31:0]
    g1[6] = 0;
    g1[7] = 0;
    i32x4 g2 = {0, 0, 0, 0};
    i32x4 g3 = {0, 0, 0, 0};
    i32x8 g4 = {0, 0, 0, 0, 0, 0, 0, 0};
    __builtin_amdgcn_tensor_load_to_lds(g0, g1, g2, g3, g4, 0);
    __builtin_amdgcn_s_wait_tensorcnt(0);
  }
  __syncthreads();
#else
  const uint4* s = (const uint4*)gsrc;
  uint4* d = (uint4*)ldst;
  unsigned int nvec = nbytes >> 4;
  for (unsigned int i = threadIdx.x; i < nvec; i += blockDim.x) d[i] = s[i];
  __syncthreads();
#endif
}

// B operand from LDS weights pre-swizzled as dst[(k/16)*(N*16) + n*16 + (k%16)].
template <int NMAT>
__device__ __forceinline__ v16bf load_B_lds(const bf16_t* Wl, int kb, int ncol, int hf) {
  const bf16_t* p0 = Wl + (kb * 2) * (NMAT * 16) + ncol * 16 + hf * 8;
  v8bf lo = *(const v8bf*)p0;
  v8bf hi = *(const v8bf*)(p0 + NMAT * 16);
  v16bf b;
#pragma unroll
  for (int i = 0; i < 8; ++i) { b[i] = lo[i]; b[i + 8] = hi[i]; }
  return b;
}

// A operand (16x32 bf16) from a row-major 16x128 bf16 LDS tile.
__device__ __forceinline__ v16bf load_A_lds(const bf16_t* ht, int m, int kb, int hf) {
  const bf16_t* p0 = ht + m * 128 + kb * 32 + hf * 8;
  v8bf lo = *(const v8bf*)p0;
  v8bf hi = *(const v8bf*)(p0 + 16);
  v16bf a;
#pragma unroll
  for (int i = 0; i < 8; ++i) { a[i] = lo[i]; a[i + 8] = hi[i]; }
  return a;
}

// A operand gathered from a global f32 row (chunks c0..c0+7 and c0+16..c0+23).
__device__ __forceinline__ v16bf gather_A_f32(const float* __restrict__ row, int c0) {
  float4 p0 = *(const float4*)(row + c0);
  float4 p1 = *(const float4*)(row + c0 + 4);
  float4 p2 = *(const float4*)(row + c0 + 16);
  float4 p3 = *(const float4*)(row + c0 + 20);
  v16bf a;
  a[0] = (bf16_t)p0.x; a[1] = (bf16_t)p0.y; a[2]  = (bf16_t)p0.z; a[3]  = (bf16_t)p0.w;
  a[4] = (bf16_t)p1.x; a[5] = (bf16_t)p1.y; a[6]  = (bf16_t)p1.z; a[7]  = (bf16_t)p1.w;
  a[8] = (bf16_t)p2.x; a[9] = (bf16_t)p2.y; a[10] = (bf16_t)p2.z; a[11] = (bf16_t)p2.w;
  a[12] = (bf16_t)p3.x; a[13] = (bf16_t)p3.y; a[14] = (bf16_t)p3.z; a[15] = (bf16_t)p3.w;
  return a;
}

// ---------------- weight prep: f32 row-major -> bf16 swizzled B layout ----------
__global__ void prep_weights(const float* __restrict__ WM1, const float* __restrict__ WM2,
                             const float* __restrict__ WM3, const float* __restrict__ WU1,
                             const float* __restrict__ WU2, const float* __restrict__ WU3,
                             bf16_t* __restrict__ dst) {
  int gid = blockIdx.x * blockDim.x + threadIdx.x;
  if (gid >= 73728) return;
  const float* src; int Nm; long long base, local;
  if (gid < 8192)       { src = WM1; Nm = 128; base = 0;     local = gid; }
  else if (gid < 24576) { src = WM2; Nm = 128; base = 8192;  local = gid - 8192; }
  else if (gid < 32768) { src = WM3; Nm = 64;  base = 24576; local = gid - 24576; }
  else if (gid < 49152) { src = WU1; Nm = 128; base = 32768; local = gid - 32768; }
  else if (gid < 65536) { src = WU2; Nm = 128; base = 49152; local = gid - 49152; }
  else                  { src = WU3; Nm = 64;  base = 65536; local = gid - 65536; }
  int k = (int)(local / Nm), n = (int)(local % Nm);
  dst[base + (long long)(k / 16) * (Nm * 16) + n * 16 + (k % 16)] = (bf16_t)src[local];
}

// ---------------- zero aggregation buffer ----------------
__global__ void zero_kernel(float* __restrict__ z, long long n) {
  long long i = ((long long)blockIdx.x * blockDim.x + threadIdx.x) * 4;
  if (i + 3 < n) {
    *(float4*)(z + i) = make_float4(0.f, 0.f, 0.f, 0.f);
  } else {
    for (long long j = i; j < n; ++j) z[j] = 0.f;
  }
}

// ---------------- edge message MLP + scatter-sum ----------------
__global__ void __launch_bounds__(TPB) edge_msg_kernel(
    const float* __restrict__ nf, const long long* __restrict__ ei,
    const float* __restrict__ bM1, const float* __restrict__ bM2,
    const float* __restrict__ bM3, const bf16_t* __restrict__ Wg,
    float* __restrict__ z, long long E, int numTiles) {
  __shared__ bf16_t Wlds[32768];          // 64 KB: WM1 @0, WM2 @8192, WM3 @24576
  __shared__ bf16_t ht[WPB][16 * 128];    // 32 KB activation bounce
  __shared__ int dstS[WPB][16];

  fill_lds_from_global(Wlds, Wg, 65536);
  const bf16_t* W1 = Wlds;
  const bf16_t* W2 = Wlds + 8192;
  const bf16_t* W3 = Wlds + 24576;

  int w = threadIdx.x >> 5;
  int lane = threadIdx.x & 31;
  int hf = lane >> 4;
  int n = lane & 15;
  int m = n;

  float bv1[8], bv2[8], bv3[4];
#pragma unroll
  for (int t = 0; t < 8; ++t) bv1[t] = bM1[t * 16 + n];
#pragma unroll
  for (int t = 0; t < 8; ++t) bv2[t] = bM2[t * 16 + n];
#pragma unroll
  for (int t = 0; t < 4; ++t) bv3[t] = bM3[t * 16 + n];

  int wid = blockIdx.x * WPB + w;
  int nwaves = gridDim.x * WPB;

  for (int tile = wid; tile < numTiles; tile += nwaves) {
    long long e = (long long)tile * 16 + m;
    long long ec = e < E ? e : E - 1;
    long long src = ei[ec];
    long long dst = ei[E + ec];
    if (hf == 0) dstS[w][m] = (int)dst;

    const float* xrow = nf + src * 64;
    v16bf a0 = gather_A_f32(xrow, hf * 8);       // K 0..31
    v16bf a1 = gather_A_f32(xrow, 32 + hf * 8);  // K 32..63

    // ---- layer 1: [16x64] @ [64x128], ReLU ----
    v8f c1[8] = {};
#pragma unroll
    for (int t = 0; t < 8; ++t) {
      c1[t] = wmma_bf16(a0, load_B_lds<128>(W1, 0, t * 16 + n, hf), c1[t]);
      c1[t] = wmma_bf16(a1, load_B_lds<128>(W1, 1, t * 16 + n, hf), c1[t]);
    }
#pragma unroll
    for (int t = 0; t < 8; ++t) {
#pragma unroll
      for (int r = 0; r < 8; ++r) {
        float v = c1[t][r] + bv1[t];
        ht[w][(r + hf * 8) * 128 + t * 16 + n] = (bf16_t)(v > 0.f ? v : 0.f);
      }
    }
    wave_sync();

    // ---- layer 2: [16x128] @ [128x128], ReLU ----
    v8f c2[8] = {};
#pragma unroll
    for (int kb = 0; kb < 4; ++kb) {
      v16bf a = load_A_lds(ht[w], m, kb, hf);
#pragma unroll
      for (int t = 0; t < 8; ++t)
        c2[t] = wmma_bf16(a, load_B_lds<128>(W2, kb, t * 16 + n, hf), c2[t]);
    }
    wave_sync();
#pragma unroll
    for (int t = 0; t < 8; ++t) {
#pragma unroll
      for (int r = 0; r < 8; ++r) {
        float v = c2[t][r] + bv2[t];
        ht[w][(r + hf * 8) * 128 + t * 16 + n] = (bf16_t)(v > 0.f ? v : 0.f);
      }
    }
    wave_sync();

    // ---- layer 3: [16x128] @ [128x64] ----
    v8f c3[4] = {};
#pragma unroll
    for (int kb = 0; kb < 4; ++kb) {
      v16bf a = load_A_lds(ht[w], m, kb, hf);
#pragma unroll
      for (int t = 0; t < 4; ++t)
        c3[t] = wmma_bf16(a, load_B_lds<64>(W3, kb, t * 16 + n, hf), c3[t]);
    }

    // ---- scatter-sum into z[dst] ----
#pragma unroll
    for (int t = 0; t < 4; ++t) {
#pragma unroll
      for (int r = 0; r < 8; ++r) {
        int row = r + hf * 8;
        long long ee = (long long)tile * 16 + row;
        if (ee < E) {
          int d = dstS[w][row];
          atomicAddF32(&z[(long long)d * 64 + t * 16 + n], c3[t][r] + bv3[t]);
        }
      }
    }
    wave_sync();
  }
}

// ---------------- node update MLP ----------------
__global__ void __launch_bounds__(TPB) node_update_kernel(
    const float* __restrict__ nf, const float* __restrict__ z,
    const float* __restrict__ bU1, const float* __restrict__ bU2,
    const float* __restrict__ bU3, const bf16_t* __restrict__ Wg,
    float* __restrict__ out, long long N, int numTiles) {
  __shared__ bf16_t Wlds[40960];          // 80 KB: WU1 @0, WU2 @16384, WU3 @32768
  __shared__ bf16_t ht[WPB][16 * 128];

  fill_lds_from_global(Wlds, Wg, 81920);
  const bf16_t* W1 = Wlds;
  const bf16_t* W2 = Wlds + 16384;
  const bf16_t* W3 = Wlds + 32768;

  int w = threadIdx.x >> 5;
  int lane = threadIdx.x & 31;
  int hf = lane >> 4;
  int n = lane & 15;
  int m = n;

  float bv1[8], bv2[8], bv3[4];
#pragma unroll
  for (int t = 0; t < 8; ++t) bv1[t] = bU1[t * 16 + n];
#pragma unroll
  for (int t = 0; t < 8; ++t) bv2[t] = bU2[t * 16 + n];
#pragma unroll
  for (int t = 0; t < 4; ++t) bv3[t] = bU3[t * 16 + n];

  int wid = blockIdx.x * WPB + w;
  int nwaves = gridDim.x * WPB;

  for (int tile = wid; tile < numTiles; tile += nwaves) {
    long long node = (long long)tile * 16 + m;
    long long ncl = node < N ? node : N - 1;
    const float* xrow = nf + ncl * 64;
    const float* zrow = z + ncl * 64;

    v16bf a[4];
    a[0] = gather_A_f32(xrow, hf * 8);        // concat: x -> K 0..63
    a[1] = gather_A_f32(xrow, 32 + hf * 8);
    a[2] = gather_A_f32(zrow, hf * 8);        //         z -> K 64..127
    a[3] = gather_A_f32(zrow, 32 + hf * 8);

    // ---- layer 1: [16x128] @ [128x128], ReLU ----
    v8f c1[8] = {};
#pragma unroll
    for (int kb = 0; kb < 4; ++kb)
#pragma unroll
      for (int t = 0; t < 8; ++t)
        c1[t] = wmma_bf16(a[kb], load_B_lds<128>(W1, kb, t * 16 + n, hf), c1[t]);
#pragma unroll
    for (int t = 0; t < 8; ++t) {
#pragma unroll
      for (int r = 0; r < 8; ++r) {
        float v = c1[t][r] + bv1[t];
        ht[w][(r + hf * 8) * 128 + t * 16 + n] = (bf16_t)(v > 0.f ? v : 0.f);
      }
    }
    wave_sync();

    // ---- layer 2: [16x128] @ [128x128], ReLU ----
    v8f c2[8] = {};
#pragma unroll
    for (int kb = 0; kb < 4; ++kb) {
      v16bf a2 = load_A_lds(ht[w], m, kb, hf);
#pragma unroll
      for (int t = 0; t < 8; ++t)
        c2[t] = wmma_bf16(a2, load_B_lds<128>(W2, kb, t * 16 + n, hf), c2[t]);
    }
    wave_sync();
#pragma unroll
    for (int t = 0; t < 8; ++t) {
#pragma unroll
      for (int r = 0; r < 8; ++r) {
        float v = c2[t][r] + bv2[t];
        ht[w][(r + hf * 8) * 128 + t * 16 + n] = (bf16_t)(v > 0.f ? v : 0.f);
      }
    }
    wave_sync();

    // ---- layer 3: [16x128] @ [128x64] ----
    v8f c3[4] = {};
#pragma unroll
    for (int kb = 0; kb < 4; ++kb) {
      v16bf a3 = load_A_lds(ht[w], m, kb, hf);
#pragma unroll
      for (int t = 0; t < 4; ++t)
        c3[t] = wmma_bf16(a3, load_B_lds<64>(W3, kb, t * 16 + n, hf), c3[t]);
    }

    // ---- store [16x64] f32 ----
#pragma unroll
    for (int t = 0; t < 4; ++t) {
#pragma unroll
      for (int r = 0; r < 8; ++r) {
        int row = r + hf * 8;
        long long nn = (long long)tile * 16 + row;
        if (nn < N) out[nn * 64 + t * 16 + n] = c3[t][r] + bv3[t];
      }
    }
    wave_sync();
  }
}

extern "C" void kernel_launch(void* const* d_in, const int* in_sizes, int n_in,
                              void* d_out, int out_size, void* d_ws, size_t ws_size,
                              hipStream_t stream) {
  const float* nf     = (const float*)d_in[0];
  const long long* ei = (const long long*)d_in[1];
  const float* WM1 = (const float*)d_in[2];  const float* bM1 = (const float*)d_in[3];
  const float* WM2 = (const float*)d_in[4];  const float* bM2 = (const float*)d_in[5];
  const float* WM3 = (const float*)d_in[6];  const float* bM3 = (const float*)d_in[7];
  const float* WU1 = (const float*)d_in[8];  const float* bU1 = (const float*)d_in[9];
  const float* WU2 = (const float*)d_in[10]; const float* bU2 = (const float*)d_in[11];
  const float* WU3 = (const float*)d_in[12]; const float* bU3 = (const float*)d_in[13];

  long long N = in_sizes[0] / 64;
  long long E = in_sizes[1] / 2;

  // workspace: [ z : N*64 f32 ][ swizzled bf16 weights : 73728 elems ]
  float* z = (float*)d_ws;
  size_t zbytes = ((size_t)N * 64 * sizeof(float) + 255) & ~(size_t)255;
  bf16_t* Wswz = (bf16_t*)((char*)d_ws + zbytes);   // message @0, update @32768

  long long zn = N * 64;
  int zBlocks = (int)((zn / 4 + TPB - 1) / TPB) + 1;
  zero_kernel<<<zBlocks, TPB, 0, stream>>>(z, zn);

  prep_weights<<<(73728 + TPB - 1) / TPB, TPB, 0, stream>>>(WM1, WM2, WM3, WU1, WU2,
                                                            WU3, Wswz);

  int eTiles = (int)((E + 15) / 16);
  int eBlocks = (eTiles + WPB - 1) / WPB;
  if (eBlocks > 1024) eBlocks = 1024;
  edge_msg_kernel<<<eBlocks, TPB, 0, stream>>>(nf, ei, bM1, bM2, bM3, Wswz, z, E,
                                               eTiles);

  int nTiles = (int)((N + 15) / 16);
  int nBlocks = (nTiles + WPB - 1) / WPB;
  if (nBlocks > 1024) nBlocks = 1024;
  node_update_kernel<<<nBlocks, TPB, 0, stream>>>(nf, z, bU1, bU2, bU3,
                                                  Wswz + 32768, (float*)d_out, N,
                                                  nTiles);
}
